// PointNetSAModule_57217554317616
// MI455X (gfx1250) — matrix operand order repaired and checked by reading, not compiled, not run on hardware
//
#include <hip/hip_runtime.h>
#include <hip/hip_bf16.h>

// ---------------------------------------------------------------------------
// PointNet++ SA module for MI455X (gfx1250, wave32, WMMA).
// bf16 activations in WMMA-B tiled layout, f32 accumulation via
// v_wmma_f32_16x16x32_bf16. GN stats fused into GEMM via shfl + atomics.
// Final max-over-K is tile-cooperative (coalesced b128 reads + LDS atomics).
// ---------------------------------------------------------------------------

typedef __attribute__((ext_vector_type(16))) __bf16 v16bf;
typedef __attribute__((ext_vector_type(8)))  float  v8f;

#define B_     8
#define N_     8192
#define S_     2048
#define KNB    32          // neighbors per center
#define NPB    65536       // S_*KNB points per batch
#define NT_B   4096        // NPB/16 n-tiles per batch

// ---- helpers --------------------------------------------------------------
__device__ __forceinline__ unsigned f2bf_pair(float a, float b) {
  union { float f; unsigned u; } x, y; x.f = a; y.f = b;
  unsigned lo = (x.u + 0x7fffu + ((x.u >> 16) & 1u)) >> 16;   // RNE
  unsigned hi = (y.u + 0x7fffu + ((y.u >> 16) & 1u)) >> 16;
  return (hi << 16) | (lo & 0xffffu);
}
__device__ __forceinline__ float bf2f(unsigned h) {
  union { unsigned u; float f; } x; x.u = h << 16; return x.f;
}
__device__ __forceinline__ float silu_f(float x) {
  return x * (1.f / (1.f + __expf(-x)));
}
// order-preserving float<->uint key for atomic max
__device__ __forceinline__ unsigned f2key(float f) {
  union { float f; unsigned u; } x; x.f = f;
  return x.u ^ ((x.u & 0x80000000u) ? 0xffffffffu : 0x80000000u);
}
__device__ __forceinline__ float key2f(unsigned k) {
  union { unsigned u; float f; } x;
  x.u = (k & 0x80000000u) ? (k ^ 0x80000000u) : ~k;
  return x.f;
}

// ---------------------------------------------------------------------------
// 1) Farthest point sampling: one 1024-thread workgroup per batch.
//    Coordinates + running min-distance live entirely in registers
//    (8 points per thread); per-iteration argmax via wave32 shfl + LDS.
// ---------------------------------------------------------------------------
__global__ __launch_bounds__(1024) void fps_kernel(const float* __restrict__ coords,
                                                   int* __restrict__ cidx) {
  const int b    = blockIdx.x;
  const int tid  = threadIdx.x;
  const int lane = tid & 31;
  const int wv   = tid >> 5;
  __shared__ float rv[32], rx[32], ry[32], rz[32];
  __shared__ int   ri[32];
  __shared__ float bc[3];

  const float* cb = coords + (size_t)b * 3 * N_;
  float px[8], py[8], pz[8], md[8];
#pragma unroll
  for (int j = 0; j < 8; ++j) {
    int i = tid + j * 1024;
    px[j] = cb[i]; py[j] = cb[N_ + i]; pz[j] = cb[2 * N_ + i];
    md[j] = 1e10f;
  }
  if (tid == 0) {
    cidx[b * S_] = 0;
    bc[0] = cb[0]; bc[1] = cb[N_]; bc[2] = cb[2 * N_];
  }
  __syncthreads();

  for (int it = 1; it < S_; ++it) {
    float lx = bc[0], ly = bc[1], lz = bc[2];
    float bestv = -1.f, bx = 0.f, by = 0.f, bz = 0.f;
    int   besti = 0x7fffffff;
#pragma unroll
    for (int j = 0; j < 8; ++j) {
      float dx = px[j] - lx, dy = py[j] - ly, dz = pz[j] - lz;
      float d = dx * dx + dy * dy + dz * dz;
      float m = fminf(md[j], d);
      md[j] = m;
      if (m > bestv) { bestv = m; besti = tid + j * 1024; bx = px[j]; by = py[j]; bz = pz[j]; }
    }
#pragma unroll
    for (int off = 16; off; off >>= 1) {
      float ov = __shfl_xor(bestv, off);
      int   oi = __shfl_xor(besti, off);
      float ox = __shfl_xor(bx, off), oy = __shfl_xor(by, off), oz = __shfl_xor(bz, off);
      if (ov > bestv || (ov == bestv && oi < besti)) {
        bestv = ov; besti = oi; bx = ox; by = oy; bz = oz;
      }
    }
    if (lane == 0) { rv[wv] = bestv; ri[wv] = besti; rx[wv] = bx; ry[wv] = by; rz[wv] = bz; }
    __syncthreads();
    if (wv == 0) {
      bestv = rv[lane]; besti = ri[lane]; bx = rx[lane]; by = ry[lane]; bz = rz[lane];
#pragma unroll
      for (int off = 16; off; off >>= 1) {
        float ov = __shfl_xor(bestv, off);
        int   oi = __shfl_xor(besti, off);
        float ox = __shfl_xor(bx, off), oy = __shfl_xor(by, off), oz = __shfl_xor(bz, off);
        if (ov > bestv || (ov == bestv && oi < besti)) {
          bestv = ov; besti = oi; bx = ox; by = oy; bz = oz;
        }
      }
      if (lane == 0) { bc[0] = bx; bc[1] = by; bc[2] = bz; cidx[b * S_ + it] = besti; }
    }
    __syncthreads();
  }
}

// ---------------------------------------------------------------------------
// 2) Gather center coordinates.
// ---------------------------------------------------------------------------
__global__ void centers_kernel(const float* __restrict__ coords,
                               const int* __restrict__ cidx,
                               float* __restrict__ centers) {
  int idx = blockIdx.x * 256 + threadIdx.x;          // (b*3+d)*S_ + s  : 49152
  int s = idx & (S_ - 1);
  int bd = idx >> 11;
  int b = bd / 3;
  centers[idx] = coords[(size_t)bd * N_ + cidx[b * S_ + s]];
}

// ---------------------------------------------------------------------------
// 3) Ball query: one wave per center, stable in-order compaction via ballot.
// ---------------------------------------------------------------------------
__global__ __launch_bounds__(256) void ballquery_kernel(const float* __restrict__ coords,
                                                        const float* __restrict__ centers,
                                                        int* __restrict__ nidx) {
  __shared__ int sbuf[8][KNB];
  const int lane = threadIdx.x & 31;
  const int wv   = threadIdx.x >> 5;
  const int cs   = blockIdx.x * 8 + wv;              // 0 .. B_*S_-1
  const int b = cs >> 11, s = cs & (S_ - 1);

  const float cx = centers[((size_t)b * 3 + 0) * S_ + s];
  const float cy = centers[((size_t)b * 3 + 1) * S_ + s];
  const float cz = centers[((size_t)b * 3 + 2) * S_ + s];
  const float* pb = coords + (size_t)b * 3 * N_;

  int cnt = 0;
  const unsigned ltm = (1u << lane) - 1u;
  for (int n0 = 0; n0 < N_ && cnt < KNB; n0 += 32) {
    int n = n0 + lane;
    float dx = pb[n] - cx, dy = pb[N_ + n] - cy, dz = pb[2 * N_ + n] - cz;
    float d2 = dx * dx + dy * dy + dz * dz;
    bool pred = d2 < 0.0625f;                         // RADIUS^2
    unsigned m = __builtin_amdgcn_ballot_w32(pred);
    int slot = cnt + __popc(m & ltm);
    if (pred && slot < KNB) sbuf[wv][slot] = n;
    cnt += __popc(m);
  }
  int v = 0;
  if (cnt > 0) {
    int first = sbuf[wv][0];
    v = (lane < cnt) ? sbuf[wv][lane] : first;
  }
  nidx[((size_t)b * S_ + s) * KNB + lane] = v;
}

// ---------------------------------------------------------------------------
// Tiled bf16 activation layout (WMMA B-fragment ready):
//   tensor[b][kt][nt] tiles of 1024 B; inside a tile, lane L owns 32 B at
//   L*32: point N = L&15, channels K = (L>>4)*16 + {2v,2v+1} for dword v.
// ---------------------------------------------------------------------------

// 4) Build layer-0 input: channels 0..2 rel-coords, 3..66 features, 67..95 zero.
__global__ void build_x0_kernel(const float* __restrict__ coords,
                                const float* __restrict__ feats,
                                const float* __restrict__ centers,
                                const int* __restrict__ nidx,
                                unsigned* __restrict__ x0) {
  int idx = blockIdx.x * 256 + threadIdx.x;          // dwords: B_*3*4096*256
  int t = idx & 255, tile = idx >> 8;
  int nt = tile & (NT_B - 1);
  int bk = tile >> 12;
  int kt = bk % 3, b = bk / 3;
  int lane = t >> 3, v = t & 7;
  int K0 = ((lane >> 4) << 4) + (v << 1);
  int c0 = kt * 32 + K0;
  int nf = nt * 16 + (lane & 15);
  int s = nf >> 5, kk = nf & 31;
  int p = nidx[((size_t)b * S_ + s) * KNB + kk];

  float vals[2];
#pragma unroll
  for (int h = 0; h < 2; ++h) {
    int c = c0 + h;
    float val;
    if (c < 3)       val = coords[((size_t)b * 3 + c) * N_ + p] -
                           centers[((size_t)b * 3 + c) * S_ + s];
    else if (c < 67) val = feats[((size_t)b * 64 + (c - 3)) * N_ + p];
    else             val = 0.f;
    vals[h] = val;
  }
  x0[idx] = f2bf_pair(vals[0], vals[1]);
}

// 5) Pack weights into WMMA A-fragment layout (contiguous k order, matching B):
//    lane L: M = L&15; K = ((L>>4)<<4) + 2*v.
__global__ void pack_w_kernel(const float* __restrict__ w, unsigned* __restrict__ dst,
                              int cin, int cout, int ktc) {
  int idx = blockIdx.x * 256 + threadIdx.x;
  int total = (cout >> 4) * ktc * 256;
  if (idx >= total) return;
  int t = idx & 255, tile = idx >> 8;
  int kt = tile % ktc, mt = tile / ktc;
  int lane = t >> 3, v = t & 7;
  int M  = lane & 15;
  int K0 = ((lane >> 4) << 4) + (v << 1);
  int co = mt * 16 + M, ci = kt * 32 + K0;
  float a = (ci     < cin) ? w[(size_t)co * cin + ci]     : 0.f;
  float b = (ci + 1 < cin) ? w[(size_t)co * cin + ci + 1] : 0.f;
  dst[idx] = f2bf_pair(a, b);
}

// 6) GEMM (+bias, +GN partial stats) with v_wmma_f32_16x16x32_bf16.
//    One wave = one 16-channel m-tile x four 16-point n-tiles, full K depth.
//    Fully templated so all inner loops unroll and A fragments sit in fixed
//    registers feeding the WMMA stream directly.
template <int KTC, int CKT_IN, int CKT_OUT, int CPG>
__global__ __launch_bounds__(128) void gemm_bias_stats(
    const char* __restrict__ xbuf, const char* __restrict__ wpack,
    const float* __restrict__ bias, char* __restrict__ ybuf,
    float* __restrict__ stats) {
  const int lane = threadIdx.x & 31;
  const int wv   = threadIdx.x >> 5;
  const int mt   = blockIdx.y;
  const int b    = blockIdx.z;
  const int m0   = mt << 4;
  const int hi   = lane >> 4;
  const int nchunk = blockIdx.x * 4 + wv;

  v16bf a[KTC];
#pragma unroll
  for (int kt = 0; kt < KTC; ++kt)
    a[kt] = *(const v16bf*)(wpack + (((size_t)(mt * KTC + kt)) << 10) + (size_t)lane * 32);

  float bv[8];
#pragma unroll
  for (int r = 0; r < 8; ++r) bv[r] = bias[m0 + hi * 8 + r];

  // D-fragment -> next-layer B-fragment store address (derived from ISA layouts)
  const size_t doff = (size_t)((lane & 15) + ((mt & 1) << 4)) * 32 + (size_t)hi * 16;
  float lsum = 0.f, lsq = 0.f;

#pragma unroll
  for (int t = 0; t < 4; ++t) {
    const int nt = nchunk * 4 + t;
    v8f c;
#pragma unroll
    for (int i = 0; i < 8; ++i) c[i] = 0.f;
#pragma unroll
    for (int kt = 0; kt < KTC; ++kt) {
      size_t toff = ((((size_t)(b * CKT_IN + kt)) << 12) + (size_t)nt) << 10;
      v16bf bb = *(const v16bf*)(xbuf + toff + (size_t)lane * 32);
      c = __builtin_amdgcn_wmma_f32_16x16x32_bf16(false, a[kt], false, bb,
                                                  (short)0, c, false, false);
    }
    unsigned o[4];
#pragma unroll
    for (int j = 0; j < 4; ++j) {
      float v0 = c[2 * j]     + bv[2 * j];
      float v1 = c[2 * j + 1] + bv[2 * j + 1];
      lsum += v0 + v1;
      lsq  += v0 * v0 + v1 * v1;
      o[j] = f2bf_pair(v0, v1);
    }
    size_t ytile = ((((size_t)(b * CKT_OUT + (mt >> 1))) << 12) + (size_t)nt) << 10;
    *(uint4*)(ybuf + ytile + doff) = make_uint4(o[0], o[1], o[2], o[3]);
  }

  // half-wave reduction (rows m0..m0+7 in lanes 0-15, m0+8..15 in 16-31)
#pragma unroll
  for (int off = 1; off < 16; off <<= 1) {
    lsum += __shfl_xor(lsum, off);
    lsq  += __shfl_xor(lsq, off);
  }
  if (CPG == 16) { lsum += __shfl_xor(lsum, 16); lsq += __shfl_xor(lsq, 16); }
  int g = (m0 + hi * 8) / CPG;
  if (lane == 0 || (CPG == 8 && lane == 16)) {
    atomicAdd(&stats[((size_t)b * 8 + g) * 2 + 0], lsum);
    atomicAdd(&stats[((size_t)b * 8 + g) * 2 + 1], lsq);
  }
}

// 7) GroupNorm + SiLU, layout-preserving (output is next GEMM's B operand).
template <int CKT, int CPG>
__global__ void gn_silu_kernel(const unsigned* __restrict__ y, unsigned* __restrict__ act,
                               const float* __restrict__ stats,
                               const float* __restrict__ gamma,
                               const float* __restrict__ beta,
                               float invcnt) {
  int idx = blockIdx.x * 256 + threadIdx.x;
  int t = idx & 255, tile = idx >> 8;
  int bk = tile >> 12;
  int kt = bk % CKT, b = bk / CKT;
  int lane = t >> 3, v = t & 7;
  int K0 = ((lane >> 4) << 4) + (v << 1);
  int c0 = kt * 32 + K0;
  int g = c0 / CPG;
  float sum = stats[((size_t)b * 8 + g) * 2 + 0];
  float sq  = stats[((size_t)b * 8 + g) * 2 + 1];
  float mu = sum * invcnt;
  float var = sq * invcnt - mu * mu;
  float rs = rsqrtf(var + 1e-5f);
  unsigned u = y[idx];
  float f0 = bf2f(u & 0xffffu), f1 = bf2f(u >> 16);
  float x0 = (f0 - mu) * rs * gamma[c0]     + beta[c0];
  float x1 = (f1 - mu) * rs * gamma[c0 + 1] + beta[c0 + 1];
  act[idx] = f2bf_pair(silu_f(x0), silu_f(x1));
}

// 8) Final: GN + SiLU + max over K neighbors -> out[b][c][s] (f32).
//    Tile-cooperative: one block per (b,s); the 32 neighbors of s are exactly
//    8 tiles (4 k-tiles x 2 n-tiles = 8 KB). Each thread reads one 32 B lane
//    chunk (2x global_load_b128, fully coalesced), applies GN+SiLU to its 16
//    channel values (single group per thread), folds into a 128-entry LDS max
//    table via order-preserving uint keys (ds_max_u32).
__global__ __launch_bounds__(256) void gn_silu_max_kernel(
    const char* __restrict__ ybuf, const float* __restrict__ stats,
    const float* __restrict__ gamma, const float* __restrict__ beta,
    float* __restrict__ out) {
  __shared__ unsigned smax[128];
  const int t  = threadIdx.x;
  const int bs = blockIdx.x;                 // b*S_ + s
  const int s  = bs & (S_ - 1), b = bs >> 11;

  if (t < 128) smax[t] = 0u;                 // below key(-inf)=0x007fffff
  __syncthreads();

  const int tile_id = t >> 5;                // 0..7
  const int lane    = t & 31;
  const int kt = tile_id >> 1, ntl = tile_id & 1;
  const int nt = s * 2 + ntl;
  const size_t base = ((((size_t)b * 4 + kt) * NT_B + nt) << 10) + (size_t)lane * 32;

  uint4 d0 = *(const uint4*)(ybuf + base);
  uint4 d1 = *(const uint4*)(ybuf + base + 16);
  unsigned dw[8] = { d0.x, d0.y, d0.z, d0.w, d1.x, d1.y, d1.z, d1.w };

  // all 16 channels of this thread fall in one 16-wide group
  const int cbase = kt * 32 + ((lane >> 4) << 4);
  const int g = cbase >> 4;
  const float invcnt = 1.f / (16.f * (float)NPB);
  const float sum = stats[((size_t)b * 8 + g) * 2 + 0];
  const float sq  = stats[((size_t)b * 8 + g) * 2 + 1];
  const float mu  = sum * invcnt;
  const float var = sq * invcnt - mu * mu;
  const float rs  = rsqrtf(var + 1e-5f);

#pragma unroll
  for (int v = 0; v < 8; ++v) {
#pragma unroll
    for (int h = 0; h < 2; ++h) {
      int c = cbase + 2 * v + h;
      float val = bf2f(h ? (dw[v] >> 16) : (dw[v] & 0xffffu));
      float xn = (val - mu) * rs * gamma[c] + beta[c];
      atomicMax(&smax[c], f2key(silu_f(xn)));
    }
  }
  __syncthreads();
  if (t < 128) {
    out[((size_t)b * 128 + t) * S_ + s] = key2f(smax[t]);
  }
}

__global__ void zero_stats_kernel(float* s) {
  if (threadIdx.x < 384) s[threadIdx.x] = 0.f;
}

// ---------------------------------------------------------------------------
// Workspace layout (bytes)
// ---------------------------------------------------------------------------
static const size_t OFF_CIDX    = 0;                          // 64 KiB
static const size_t OFF_NIDX    = (size_t)1 << 20;            // 2 MiB
static const size_t OFF_CENTERS = (size_t)4 << 20;            // 192 KiB
static const size_t OFF_STATS   = (size_t)5 << 20;            // 1.5 KiB
static const size_t OFF_WP0     = (size_t)6 << 20;
static const size_t OFF_WP1     = ((size_t)6 << 20) + 16384;
static const size_t OFF_WP2     = ((size_t)6 << 20) + 32768;
static const size_t OFF_X0      = (size_t)8 << 20;            // 96 MiB (x0 / act1)
static const size_t OFF_ACT     = OFF_X0 + ((size_t)32 << 20);// 64 MiB (overlaps x0 tail)
static const size_t OFF_Y       = (size_t)104 << 20;          // 128 MiB

extern "C" void kernel_launch(void* const* d_in, const int* in_sizes, int n_in,
                              void* d_out, int out_size, void* d_ws, size_t ws_size,
                              hipStream_t stream) {
  (void)in_sizes; (void)n_in; (void)out_size; (void)ws_size;
  const float* coords = (const float*)d_in[0];
  const float* feats  = (const float*)d_in[1];
  const float* w0 = (const float*)d_in[2],  *b0 = (const float*)d_in[3];
  const float* g0 = (const float*)d_in[4],  *be0 = (const float*)d_in[5];
  const float* w1 = (const float*)d_in[6],  *b1 = (const float*)d_in[7];
  const float* g1 = (const float*)d_in[8],  *be1 = (const float*)d_in[9];
  const float* w2 = (const float*)d_in[10], *b2 = (const float*)d_in[11];
  const float* g2 = (const float*)d_in[12], *be2 = (const float*)d_in[13];
  float* out = (float*)d_out;

  char* ws = (char*)d_ws;
  int*      cidx    = (int*)(ws + OFF_CIDX);
  int*      nidx    = (int*)(ws + OFF_NIDX);
  float*    centers = (float*)(ws + OFF_CENTERS);
  float*    stats   = (float*)(ws + OFF_STATS);     // [3][B][8][2]
  unsigned* wp0     = (unsigned*)(ws + OFF_WP0);
  unsigned* wp1     = (unsigned*)(ws + OFF_WP1);
  unsigned* wp2     = (unsigned*)(ws + OFF_WP2);
  char*     x0buf   = ws + OFF_X0;
  char*     actbuf  = ws + OFF_ACT;
  char*     ybuf    = ws + OFF_Y;

  zero_stats_kernel<<<1, 512, 0, stream>>>(stats);
  fps_kernel<<<B_, 1024, 0, stream>>>(coords, cidx);
  centers_kernel<<<192, 256, 0, stream>>>(coords, cidx, centers);
  ballquery_kernel<<<(B_ * S_) / 8, 256, 0, stream>>>(coords, centers, nidx);

  pack_w_kernel<<<12, 256, 0, stream>>>(w0, wp0, 67, 64, 3);
  pack_w_kernel<<<8,  256, 0, stream>>>(w1, wp1, 64, 64, 2);
  pack_w_kernel<<<16, 256, 0, stream>>>(w2, wp2, 64, 128, 2);

  build_x0_kernel<<<98304, 256, 0, stream>>>(coords, feats, centers, nidx,
                                             (unsigned*)x0buf);

  const float inv8 = 1.f / (8.f * (float)NPB);

  // Layer 0: x0(96ch) -> y(64ch)
  gemm_bias_stats<3, 3, 2, 8><<<dim3(256, 4, B_), 128, 0, stream>>>(
      x0buf, (const char*)wp0, b0, ybuf, stats + 0);
  gn_silu_kernel<2, 8><<<65536, 256, 0, stream>>>((const unsigned*)ybuf,
                                                  (unsigned*)actbuf,
                                                  stats + 0, g0, be0, inv8);
  // Layer 1: act(64ch) -> y(64ch)
  gemm_bias_stats<2, 2, 2, 8><<<dim3(256, 4, B_), 128, 0, stream>>>(
      actbuf, (const char*)wp1, b1, ybuf, stats + 128);
  gn_silu_kernel<2, 8><<<65536, 256, 0, stream>>>((const unsigned*)ybuf,
                                                  (unsigned*)x0buf,
                                                  stats + 128, g1, be1, inv8);
  // Layer 2: act1(64ch, in x0 region) -> y(128ch)
  gemm_bias_stats<2, 2, 4, 16><<<dim3(256, 8, B_), 128, 0, stream>>>(
      x0buf, (const char*)wp2, b2, ybuf, stats + 256);

  gn_silu_max_kernel<<<B_ * S_, 256, 0, stream>>>(ybuf, stats + 256, g2, be2, out);
}
